// AttentionBackward_4595615007291
// MI455X (gfx1250) — matrix operand order repaired and checked by reading, not compiled, not run on hardware
//
#include <hip/hip_runtime.h>
#include <hip/hip_bf16.h>

typedef __attribute__((ext_vector_type(16))) _Float16 v16h;
typedef __attribute__((ext_vector_type(8)))  _Float16 v8h;
typedef __attribute__((ext_vector_type(8)))  float    v8f;

__device__ __forceinline__ v16h cat8(v8h lo, v8h hi) {
  return __builtin_shufflevector(lo, hi, 0,1,2,3,4,5,6,7,8,9,10,11,12,13,14,15);
}

// Problem constants
constexpr int B_  = 4;
constexpr int S_  = 1088;
constexpr int H_  = 1024;
constexpr int NH_ = 16;
constexpr int HD_ = 64;
constexpr int P_  = 1088;
constexpr int SEQ_ = 1024;
constexpr int NBp_ = 64;      // num prefixes
constexpr int NTOK = B_ * S_; // 4352 tokens
constexpr int JW_  = S_ / 32; // 34 mask words per row

// Workspace layout (bytes); every size is a multiple of 256
constexpr size_t SZ_T16  = (size_t)NTOK * NH_ * HD_ * 2;  // 8,912,896
constexpr size_t OFF_TQ  = 0;
constexpr size_t OFF_TK  = OFF_TQ + SZ_T16;
constexpr size_t OFF_TV  = OFF_TK + SZ_T16;
constexpr size_t OFF_Q16 = OFF_TV + SZ_T16;
constexpr size_t OFF_K16 = OFF_Q16 + SZ_T16;
constexpr size_t OFF_VT  = OFF_K16 + SZ_T16;                    // V transposed [b][h][d][s] f16
constexpr size_t OFF_PB  = OFF_VT + SZ_T16;                     // [B,NH,S] f32
constexpr size_t SZ_PB   = (size_t)B_ * NH_ * S_ * 4;
constexpr size_t OFF_AB  = OFF_PB + SZ_PB;                      // [B,JW,S] u32
constexpr size_t SZ_AB   = (size_t)B_ * S_ * JW_ * 4;
constexpr size_t OFF_NBM = OFF_AB + SZ_AB;                      // [JW,S] u32
constexpr size_t SZ_NBM  = (size_t)S_ * JW_ * 4;
constexpr size_t OFF_CTX = OFF_NBM + SZ_NBM;                    // [tok,NH,HD] f32
constexpr size_t SZ_CTX  = (size_t)NTOK * NH_ * HD_ * 4;
constexpr size_t OFF_OUTB = OFF_CTX + SZ_CTX;                   // [tok,H] f32
constexpr size_t SZ_OUTB  = (size_t)NTOK * H_ * 4;
constexpr size_t OFF_WT   = OFF_OUTB + SZ_OUTB;                 // f16 W^T [p][i][e][d]
// total ~90.5 MB

// ---------------------------------------------------------------------------
// 1) Head mixing: t[tok,i,d] = sum_j mix[d,i,j] * x[tok,j,d], for q/k/v (f16 out)
// ---------------------------------------------------------------------------
__global__ void headmix_kernel(const float* __restrict__ hidden,
                               const float* __restrict__ qh,
                               const float* __restrict__ kh,
                               const float* __restrict__ vh,
                               _Float16* __restrict__ tq,
                               _Float16* __restrict__ tk,
                               _Float16* __restrict__ tv) {
  __shared__ float xs[8 * 1024];
  const int tid = threadIdx.x;
  const long tok0 = (long)blockIdx.x * 8;
  const float* src = hidden + tok0 * H_;
  for (int idx = tid; idx < 8 * 1024; idx += 256) xs[idx] = src[idx];
  __syncthreads();
  const int d  = tid & 63;
  const int i0 = tid >> 6; // 0..3
  const float* mixes[3] = {qh, kh, vh};
  _Float16*    outs[3]  = {tq, tk, tv};
  for (int p = 0; p < 3; ++p) {
    const float* mx = mixes[p];
    _Float16* op = outs[p];
    for (int ii = 0; ii < 4; ++ii) {
      const int i = ii * 4 + i0;
      float w[16];
#pragma unroll
      for (int j = 0; j < 16; ++j) w[j] = mx[(size_t)d * 256 + i * 16 + j];
      for (int tt = 0; tt < 8; ++tt) {
        const float* xp = xs + tt * 1024 + d;
        float acc = 0.f;
#pragma unroll
        for (int j = 0; j < 16; ++j) acc += w[j] * xp[j * 64];
        op[((tok0 + tt) * NH_ + i) * HD_ + d] = (_Float16)acc;
      }
    }
  }
}

// ---------------------------------------------------------------------------
// 1b) Weight transpose+convert: wt[p][i][e][d] = (f16) W_p[i][d][e]
// ---------------------------------------------------------------------------
__global__ void wt16_kernel(const float* __restrict__ qa,
                            const float* __restrict__ ka,
                            const float* __restrict__ va,
                            _Float16* __restrict__ wt) {
  const long idx = (long)blockIdx.x * 256 + threadIdx.x;  // 3*16*64*64 total
  const int d = (int)(idx & 63);
  const int e = (int)((idx >> 6) & 63);
  const int i = (int)((idx >> 12) & 15);
  const int p = (int)(idx >> 16);
  const float* W = (p == 0) ? qa : (p == 1) ? ka : va;
  wt[idx] = (_Float16)W[((size_t)i * 64 + d) * 64 + e];
}

// ---------------------------------------------------------------------------
// 2a) Per-head 64x64 projection via WMMA for Q and K (uniform store path).
//     One wave per (16-token tile, head, proj in {q,k}).
// ---------------------------------------------------------------------------
__global__ void headproj_qk_kernel(const _Float16* __restrict__ tq,
                                   const _Float16* __restrict__ tk,
                                   const _Float16* __restrict__ wt,
                                   _Float16* __restrict__ q16,
                                   _Float16* __restrict__ k16) {
  const int tid  = threadIdx.x;
  const int lane = tid & 31;
  const int gw   = blockIdx.x * 8 + (tid >> 5);
  const int mt   = gw % (NTOK / 16);          // 0..271
  const int hd   = (gw / (NTOK / 16)) % NH_;  // head
  const int p    = gw / ((NTOK / 16) * NH_);  // 0..1
  const _Float16* tsrc = (p == 0) ? tq : tk;
  _Float16*       dst  = (p == 0) ? q16 : k16;
  const int lm = lane & 15, lh = lane >> 4;
  const long tok0 = (long)mt * 16;

  const _Float16* tp = tsrc + ((size_t)(tok0 + lm) * NH_ + hd) * HD_;
  const v16h a0 = cat8(*(const v8h*)(tp + lh * 16), *(const v8h*)(tp + 8 + lh * 16));
  const v16h a1 = cat8(*(const v8h*)(tp + 32 + lh * 16), *(const v8h*)(tp + 40 + lh * 16));

  const _Float16* wbase = wt + (size_t)(p * NH_ + hd) * HD_ * HD_;  // [e][d]
#pragma unroll
  for (int nt = 0; nt < 4; ++nt) {
    const int e = nt * 16 + lm;
    const _Float16* wr = wbase + (size_t)e * HD_;
    const v16h b0 = *(const v16h*)(wr + lh * 16);
    const v16h b1 = *(const v16h*)(wr + 32 + lh * 16);
    v8f acc = {};
    acc = __builtin_amdgcn_wmma_f32_16x16x32_f16(false, a0, false, b0, (short)0, acc, false, false);
    acc = __builtin_amdgcn_wmma_f32_16x16x32_f16(false, a1, false, b1, (short)0, acc, false, false);
#pragma unroll
    for (int r = 0; r < 8; ++r) {
      const long tok = tok0 + 8 * lh + r;
      dst[((size_t)tok * NH_ + hd) * HD_ + e] = (_Float16)acc[r];
    }
  }
}

// ---------------------------------------------------------------------------
// 2b) Per-head projection for V, stored transposed [b][h][d][s] (uniform path).
// ---------------------------------------------------------------------------
__global__ void headproj_v_kernel(const _Float16* __restrict__ tv,
                                  const _Float16* __restrict__ wt,
                                  _Float16* __restrict__ vT) {
  const int tid  = threadIdx.x;
  const int lane = tid & 31;
  const int gw   = blockIdx.x * 8 + (tid >> 5);
  const int mt   = gw % (NTOK / 16);          // 0..271
  const int hd   = gw / (NTOK / 16);          // head
  const int lm = lane & 15, lh = lane >> 4;
  const long tok0 = (long)mt * 16;

  const _Float16* tp = tv + ((size_t)(tok0 + lm) * NH_ + hd) * HD_;
  const v16h a0 = cat8(*(const v8h*)(tp + lh * 16), *(const v8h*)(tp + 8 + lh * 16));
  const v16h a1 = cat8(*(const v8h*)(tp + 32 + lh * 16), *(const v8h*)(tp + 40 + lh * 16));

  const _Float16* wbase = wt + (size_t)(2 * NH_ + hd) * HD_ * HD_;  // v weights, [e][d]
  const long bb = tok0 / S_;
  const long s0 = tok0 - bb * S_;
#pragma unroll
  for (int nt = 0; nt < 4; ++nt) {
    const int e = nt * 16 + lm;
    const _Float16* wr = wbase + (size_t)e * HD_;
    const v16h b0 = *(const v16h*)(wr + lh * 16);
    const v16h b1 = *(const v16h*)(wr + 32 + lh * 16);
    v8f acc = {};
    acc = __builtin_amdgcn_wmma_f32_16x16x32_f16(false, a0, false, b0, (short)0, acc, false, false);
    acc = __builtin_amdgcn_wmma_f32_16x16x32_f16(false, a1, false, b1, (short)0, acc, false, false);
    _Float16* vrow = vT + (((size_t)bb * NH_ + hd) * HD_ + e) * S_ + s0 + 8 * lh;
#pragma unroll
    for (int r = 0; r < 8; ++r) {
      vrow[r] = (_Float16)acc[r];
    }
  }
}

// ---------------------------------------------------------------------------
// 3) Positional bias: pbias[b,h,j] = sum_p pos[b,j,p] * pw[h,p]
// ---------------------------------------------------------------------------
__global__ void pbias_kernel(const float* __restrict__ pos,
                             const float* __restrict__ pw,
                             float* __restrict__ pbias) {
  __shared__ float ps[P_];
  __shared__ float red[256];
  const long g = blockIdx.x; // b*S + j
  const int tid = threadIdx.x;
  const float* prow = pos + g * P_;
  for (int c = tid; c < P_; c += 256) ps[c] = prow[c];
  __syncthreads();
  const int h = tid >> 4, part = tid & 15;
  float acc = 0.f;
  for (int p = part; p < P_; p += 16) acc += ps[p] * pw[(size_t)h * P_ + p];
  red[tid] = acc;
  __syncthreads();
  if (tid < 16) {
    float s = 0.f;
#pragma unroll
    for (int q = 0; q < 16; ++q) s += red[tid * 16 + q];
    const long b = g / S_, j = g % S_;
    pbias[((size_t)b * NH_ + tid) * S_ + j] = s;
  }
}

// ---------------------------------------------------------------------------
// 4) Mask bit-packing ([b][w][i] / [w][i] layouts: rows contiguous)
// ---------------------------------------------------------------------------
__global__ void attnbits_kernel(const float* __restrict__ am, unsigned* __restrict__ ab) {
  const long idx = (long)blockIdx.x * 256 + threadIdx.x;
  if (idx >= (long)B_ * S_ * JW_) return;
  const int w = (int)(idx % JW_);
  const int i = (int)((idx / JW_) % S_);
  const int b = (int)(idx / ((long)JW_ * S_));
  const float* row = am + ((size_t)b * S_ + i) * S_;
  unsigned bits = 0;
  for (int bit = 0; bit < 32; ++bit) {
    const int j = w * 32 + bit;
    const bool ok = (j < NBp_) || (row[j] >= -0.5f);
    bits |= (unsigned)ok << bit;
  }
  ab[((size_t)b * JW_ + w) * S_ + i] = bits;
}

__global__ void normbits_kernel(const float* __restrict__ icl, unsigned* __restrict__ nb) {
  const long idx = (long)blockIdx.x * 256 + threadIdx.x;
  if (idx >= (long)S_ * JW_) return;
  const int w = (int)(idx % JW_);
  const int i = (int)(idx / JW_);
  unsigned bits = 0;
  for (int bit = 0; bit < 32; ++bit) {
    const int j = w * 32 + bit;
    bool ok;
    if (j < NBp_) ok = false;
    else if (i < NBp_) ok = true;
    else {
      const int ii = i - NBp_, jj = j - NBp_;
      if (ii < jj) ok = true;  // above diagonal: tril zeroes the block
      else {
        float c = 1.f - icl[(size_t)ii * SEQ_ + jj];
        c = fminf(fmaxf(c, 0.f), 1.f);
        ok = (rintf(c) < 0.5f);
      }
    }
    bits |= (unsigned)ok << bit;
  }
  nb[(size_t)w * S_ + i] = bits;
}

// ---------------------------------------------------------------------------
// 5) Fused linear attention. One wave per (b, head, 16-query tile); streams
//    32-key tiles; scores + ctx via v_wmma_f32_16x16x32_f16; denominator
//    accumulated per D-fragment row; P tile re-fragmented through LDS.
//    All fragments are contiguous 16/32-byte loads.
// ---------------------------------------------------------------------------
__global__ void attn_kernel(const _Float16* __restrict__ q16,
                            const _Float16* __restrict__ k16,
                            const _Float16* __restrict__ vT,
                            const float* __restrict__ pbias,
                            const unsigned* __restrict__ abits,
                            const unsigned* __restrict__ nbits,
                            float* __restrict__ ctx) {
  __shared__ _Float16 ptile[8][16 * 32];
  const int tid = threadIdx.x, lane = tid & 31, wv = tid >> 5;
  const int gw = blockIdx.x * 8 + wv;
  const int it = gw % (S_ / 16);
  const int hq = (gw / (S_ / 16)) % NH_;
  const int b  = gw / ((S_ / 16) * NH_);
  const int i0 = it * 16;
  const int lm = lane & 15, lh = lane >> 4;

  // Q A-fragments (rows = queries, K = d)
  const _Float16* qp = q16 + (((size_t)b * S_ + i0 + lm) * NH_ + hq) * HD_;
  const v16h aq0 = cat8(*(const v8h*)(qp + lh * 16), *(const v8h*)(qp + 8 + lh * 16));
  const v16h aq1 = cat8(*(const v8h*)(qp + 32 + lh * 16), *(const v8h*)(qp + 40 + lh * 16));

  float dn[8];
#pragma unroll
  for (int r = 0; r < 8; ++r) dn[r] = 1e-6f;
  v8f cc[4];
#pragma unroll
  for (int nt = 0; nt < 4; ++nt) cc[nt] = (v8f){};
  const float* pbrow = pbias + ((size_t)b * NH_ + hq) * S_;

  for (int jt = 0; jt < JW_; ++jt) {
    const int j0 = jt * 32;
    // mask words for this wave's 8 rows (contiguous in new layout)
    const unsigned* ab8 = abits + ((size_t)b * JW_ + jt) * S_ + i0 + 8 * lh;
    const unsigned* nb8 = nbits + (size_t)jt * S_ + i0 + 8 * lh;
    const uint4 a0v = *(const uint4*)ab8;
    const uint4 a1v = *(const uint4*)(ab8 + 4);
    const uint4 n0v = *(const uint4*)nb8;
    const uint4 n1v = *(const uint4*)(nb8 + 4);
    const unsigned aw[8] = {a0v.x, a0v.y, a0v.z, a0v.w, a1v.x, a1v.y, a1v.z, a1v.w};
    const unsigned nw[8] = {n0v.x, n0v.y, n0v.z, n0v.w, n1v.x, n1v.y, n1v.z, n1v.w};

#pragma unroll
    for (int sub = 0; sub < 2; ++sub) {
      const int j0s = j0 + sub * 16;
      const _Float16* kp = k16 + (((size_t)b * S_ + j0s + lm) * NH_ + hq) * HD_;
      const v16h b0 = *(const v16h*)(kp + lh * 16);
      const v16h b1 = *(const v16h*)(kp + 32 + lh * 16);
      v8f s = {};
      s = __builtin_amdgcn_wmma_f32_16x16x32_f16(false, aq0, false, b0, (short)0, s, false, false);
      s = __builtin_amdgcn_wmma_f32_16x16x32_f16(false, aq1, false, b1, (short)0, s, false, false);
      const float pb = pbrow[j0s + lm];
      const int jb = sub * 16 + lm;
#pragma unroll
      for (int r = 0; r < 8; ++r) {
        float val = s[r] * 0.125f + pb;           // /sqrt(64) then +bias
        val *= (float)((aw[r] >> jb) & 1u);       // attn mask (zeroing)
        dn[r] += fabsf(val) * (float)((nw[r] >> jb) & 1u);
        ptile[wv][(8 * lh + r) * 32 + jb] = (_Float16)val;
      }
    }
    // Re-fragment P tile as A operand (K = 32 keys)
    const _Float16* pt = &ptile[wv][lm * 32];
    const v16h ap = cat8(*(const v8h*)(pt + lh * 16), *(const v8h*)(pt + 8 + lh * 16));
#pragma unroll
    for (int nt = 0; nt < 4; ++nt) {
      const _Float16* vr = vT + (((size_t)b * NH_ + hq) * HD_ + nt * 16 + lm) * S_ + j0;
      const v16h bv = *(const v16h*)(vr + lh * 16);
      cc[nt] = __builtin_amdgcn_wmma_f32_16x16x32_f16(false, ap, false, bv, (short)0, cc[nt], false, false);
    }
    // prefetch next key tile rows (global_prefetch_b8)
    if (jt + 1 < JW_) {
      __builtin_prefetch(k16 + (((size_t)b * S_ + j0 + 32 + lm) * NH_ + hq) * HD_, 0, 0);
      __builtin_prefetch(k16 + (((size_t)b * S_ + j0 + 48 + lm) * NH_ + hq) * HD_, 0, 0);
    }
  }
  // Divide by denominator, store ctx
#pragma unroll
  for (int nt = 0; nt < 4; ++nt) {
#pragma unroll
    for (int r = 0; r < 8; ++r) {
      const float o = cc[nt][r] / dn[r];
      ctx[(((size_t)b * S_ + i0 + 8 * lh + r) * NH_ + hq) * HD_ + nt * 16 + lm] = o;
    }
  }
}

// ---------------------------------------------------------------------------
// 6) c_proj head mixing + residual slab
// ---------------------------------------------------------------------------
__global__ void cproj_kernel(const float* __restrict__ ctx,
                             const float* __restrict__ cpw,
                             const float* __restrict__ hidden,
                             float* __restrict__ outb) {
  __shared__ float cs[8 * 1024];
  const int tid = threadIdx.x;
  const long g0 = (long)blockIdx.x * 8;
  const float* src = ctx + g0 * (NH_ * HD_);
  for (int idx = tid; idx < 8 * 1024; idx += 256) cs[idx] = src[idx];
  __syncthreads();
  const int d = tid & 63, i0 = tid >> 6;
  for (int ii = 0; ii < 4; ++ii) {
    const int i = ii * 4 + i0;
    float w[16];
#pragma unroll
    for (int j = 0; j < 16; ++j) w[j] = cpw[(size_t)d * 256 + i * 16 + j];
    const int o = i * 64 + d;
    for (int tt = 0; tt < 8; ++tt) {
      const float* cp = cs + tt * 1024 + d;
      float acc = 0.f;
#pragma unroll
      for (int j = 0; j < 16; ++j) acc += w[j] * cp[j * 64];
      const long g = g0 + tt;
      const int s = (int)(g % S_);
      if (s >= NBp_ && o >= 256) acc += hidden[g * H_ + o];  // MI..MI+3*DIN = 256..1023
      outb[g * H_ + o] = acc;
    }
  }
}

// ---------------------------------------------------------------------------
// 7) Gating: sigmoid(w.h + u.out + v.pos + b) -> blend
// ---------------------------------------------------------------------------
__global__ void gate_kernel(const float* __restrict__ hidden,
                            const float* __restrict__ outb,
                            const float* __restrict__ pos,
                            const float* __restrict__ gw, const float* __restrict__ gu,
                            const float* __restrict__ gv, const float* __restrict__ gwb,
                            const float* __restrict__ gub, const float* __restrict__ gvb,
                            float* __restrict__ out) {
  __shared__ float r0[256], r1[256];
  __shared__ float gbc[2];
  const int tid = threadIdx.x;
  const long g = blockIdx.x;
  const float* hrow = hidden + g * H_;
  const float* orow = outb + g * H_;
  const float* prow = pos + g * P_;
  float p0 = 0.f, p1 = 0.f;
  for (int c = tid; c < H_; c += 256) {
    const float hv = hrow[c], ov = orow[c];
    p0 += hv * gw[c] + ov * gu[c];
    p1 += hv * gw[H_ + c] + ov * gu[H_ + c];
  }
  for (int c = tid; c < P_; c += 256) {
    const float pv = prow[c];
    p0 += pv * gv[c];
    p1 += pv * gv[P_ + c];
  }
  r0[tid] = p0; r1[tid] = p1;
  __syncthreads();
  for (int st = 128; st > 0; st >>= 1) {
    if (tid < st) { r0[tid] += r0[tid + st]; r1[tid] += r1[tid + st]; }
    __syncthreads();
  }
  if (tid == 0) {
    const float l0 = r0[0] + gwb[0] + gub[0] + gvb[0];
    const float l1 = r1[0] + gwb[1] + gub[1] + gvb[1];
    gbc[0] = 1.f / (1.f + __expf(-l0));
    gbc[1] = 1.f / (1.f + __expf(-l1));
  }
  __syncthreads();
  const float g0v = gbc[0], g1v = gbc[1];
  for (int c = tid; c < H_; c += 256) out[g * H_ + c] = g0v * hrow[c] + g1v * orow[c];
}

// ---------------------------------------------------------------------------
extern "C" void kernel_launch(void* const* d_in, const int* in_sizes, int n_in,
                              void* d_out, int out_size, void* d_ws, size_t ws_size,
                              hipStream_t stream) {
  const float* hidden   = (const float*)d_in[0];
  const float* pos      = (const float*)d_in[1];
  const float* amask    = (const float*)d_in[2];
  const float* icl      = (const float*)d_in[3];
  const float* q_attn   = (const float*)d_in[4];
  const float* q_head   = (const float*)d_in[5];
  const float* k_attn   = (const float*)d_in[6];
  const float* k_head   = (const float*)d_in[7];
  const float* v_attn   = (const float*)d_in[8];
  const float* v_head   = (const float*)d_in[9];
  const float* c_proj   = (const float*)d_in[10];
  const float* p_attn_w = (const float*)d_in[11];
  const float* gw  = (const float*)d_in[12];
  const float* gu  = (const float*)d_in[13];
  const float* gv  = (const float*)d_in[14];
  const float* gwb = (const float*)d_in[15];
  const float* gub = (const float*)d_in[16];
  const float* gvb = (const float*)d_in[17];
  float* out = (float*)d_out;

  char* ws = (char*)d_ws;
  _Float16* tq  = (_Float16*)(ws + OFF_TQ);
  _Float16* tk  = (_Float16*)(ws + OFF_TK);
  _Float16* tv  = (_Float16*)(ws + OFF_TV);
  _Float16* q16 = (_Float16*)(ws + OFF_Q16);
  _Float16* k16 = (_Float16*)(ws + OFF_K16);
  _Float16* vT  = (_Float16*)(ws + OFF_VT);
  float*    pb  = (float*)(ws + OFF_PB);
  unsigned* ab  = (unsigned*)(ws + OFF_AB);
  unsigned* nbm = (unsigned*)(ws + OFF_NBM);
  float*    ctx = (float*)(ws + OFF_CTX);
  float*    outb = (float*)(ws + OFF_OUTB);
  _Float16* wt  = (_Float16*)(ws + OFF_WT);

  headmix_kernel<<<NTOK / 8, 256, 0, stream>>>(hidden, q_head, k_head, v_head, tq, tk, tv);
  wt16_kernel<<<(3 * NH_ * HD_ * HD_) / 256, 256, 0, stream>>>(q_attn, k_attn, v_attn, wt);
  headproj_qk_kernel<<<(NTOK / 16) * NH_ * 2 / 8, 256, 0, stream>>>(tq, tk, wt, q16, k16);
  headproj_v_kernel<<<(NTOK / 16) * NH_ / 8, 256, 0, stream>>>(tv, wt, vT);
  pbias_kernel<<<NTOK, 256, 0, stream>>>(pos, p_attn_w, pb);
  attnbits_kernel<<<((long)B_ * S_ * JW_ + 255) / 256, 256, 0, stream>>>(amask, ab);
  normbits_kernel<<<((long)S_ * JW_ + 255) / 256, 256, 0, stream>>>(icl, nbm);
  attn_kernel<<<B_ * NH_ * (S_ / 16) / 8, 256, 0, stream>>>(q16, k16, vT, pb, ab, nbm, ctx);
  cproj_kernel<<<NTOK / 8, 256, 0, stream>>>(ctx, c_proj, hidden, outb);
  gate_kernel<<<NTOK, 256, 0, stream>>>(hidden, outb, pos, gw, gu, gv, gwb, gub, gvb, out);
}
